// SConv2dAvg_51342039056477
// MI455X (gfx1250) — compile-verified, matrix operand order checked by reading, not run on hardware
//
#include <hip/hip_runtime.h>

typedef __attribute__((ext_vector_type(16))) __bf16        v16bf;
typedef __attribute__((ext_vector_type(8)))  float         v8f;
typedef __attribute__((ext_vector_type(8)))  unsigned int  v8u;

#define KC  64   // K-chunk = one 3x3 tap across all 64 input channels
#define KCP 72   // padded LDS row stride in bf16 elems (144 B: keeps 16 B segs aligned)

// ---------------- pre-pass: weight fp32 [Cout][Cin][3][3] -> bf16 [m][t][cin] ----------------
__global__ __launch_bounds__(256)
void weight_permute_bf16(const float* __restrict__ w, __bf16* __restrict__ ws) {
  int i = blockIdx.x * 256 + threadIdx.x;      // i = m*576 + t*64 + cin
  if (i < 64 * 576) {
    int m   = i / 576;
    int r   = i - m * 576;
    int t   = r >> 6;                          // tap index 0..8
    int cin = r & 63;
    ws[i] = (__bf16)w[(m * 64 + cin) * 9 + t];
  }
}

// ---------------- main implicit-GEMM conv ----------------
__global__ __launch_bounds__(256)
void sconv2d_wmma_kernel(const float* __restrict__ input,
                         const __bf16* __restrict__ wsW,   // permuted bf16 weights [m][t][cin]
                         const float* __restrict__ bias,
                         const int*   __restrict__ selh,
                         const int*   __restrict__ selw,
                         float*       __restrict__ out)
{
  constexpr int Cin = 64, Cout = 64, H = 128, W = 128;
  constexpr int OH = 63, OW = 63;
  constexpr int NT = 32;        // output pixels per block
  constexpr int KW = 576;       // Cin*9

  __shared__ __bf16 Wlds[Cout * KCP]; // [m][cin]  (one tap per chunk)
  __shared__ __bf16 Plds[NT * KCP];   // [j][cin]
  __shared__ int    ihw[NT];          // ih0*W + iw0 per pixel

  const int tid = threadIdx.x;
  const int ow0 = blockIdx.x * NT;
  const int y   = blockIdx.y;
  const int b   = blockIdx.z;

  if (tid < NT) {
    int x  = ow0 + tid;
    int xc = (x < OW) ? x : (OW - 1);          // clamp edge column (store is guarded)
    int ih0 = 2 * y  + selh[y * OW + xc];
    int iw0 = 2 * xc + selw[y * OW + xc];
    ihw[tid] = ih0 * W + iw0;
  }
  __syncthreads();

  const int wave = tid >> 5;
  const int lane = tid & 31;
  const int m0   = (wave & 3) * 16;   // Cout sub-tile
  const int j0   = (wave >> 2) * 16;  // pixel sub-tile
  const int row  = lane & 15;
  const int hi8  = (lane >> 4) * 8;   // lanes 16..31 hold K+8 per ISA A/B layout

  v8f acc = {};

  const float* inB = input + (size_t)b * Cin * H * W;

  for (int kc = 0; kc < 9; ++kc) {    // one 3x3 tap per chunk; k = kc*64 + cin
    const int dh = kc / 3;            // uniform per chunk (SALU)
    const int dw = kc - dh * 3;

    // ---- async-copy bf16 weight chunk [64m][64cin] global -> LDS (ASYNCcnt) ----
    #pragma unroll
    for (int i = 0; i < 2; ++i) {
      int tr  = i * 256 + tid;        // 512 x 16B transfers
      int m   = tr >> 3;
      int seg = tr & 7;
      unsigned      ldsb = (unsigned)(size_t)&Wlds[m * KCP + seg * 8];
      const __bf16* gsrc = wsW + m * KW + kc * 64 + seg * 8;
      asm volatile("global_load_async_to_lds_b128 %0, %1, off"
                   :: "v"(ldsb), "v"(gsrc) : "memory");
    }

    // ---- gather patch chunk: Plds[j][cin] = bf16(input[b][cin][ih0+dh][iw0+dw]) ----
    {
      const int cin = tid & 63;       // fixed per thread
      const int jb  = tid >> 6;
      const float* src = inB + cin * (H * W) + dh * W + dw;
      #pragma unroll
      for (int i = 0; i < 8; ++i) {
        int j = jb + i * 4;
        Plds[j * KCP + cin] = (__bf16)src[ihw[j]];
      }
    }

    asm volatile("s_wait_asynccnt 0x0" ::: "memory");
    __syncthreads();

    // ---- two K=32 WMMA steps per staged chunk ----
    #pragma unroll
    for (int ks = 0; ks < 2; ++ks) {
      int kb = ks * 32;
      v8u ab, bb;
      #pragma unroll
      for (int v = 0; v < 8; ++v) {
        // 16-bit A/B 16x32 layout: VGPR v holds K pair {2v,..}(v<4)/{2v+8,..}(v>=4), +8 for hi lanes
        int kk = kb + ((v < 4) ? 2 * v : 2 * v + 8) + hi8;
        ab[v] = *(const unsigned int*)&Wlds[(m0 + row) * KCP + kk];
        bb[v] = *(const unsigned int*)&Plds[(j0 + row) * KCP + kk];
      }
      v16bf a  = __builtin_bit_cast(v16bf, ab);
      v16bf bm = __builtin_bit_cast(v16bf, bb);
      acc = __builtin_amdgcn_wmma_f32_16x16x32_bf16(false, a, false, bm,
                                                    (short)0, acc, false, false);
    }
    __syncthreads();
  }

  // C/D layout: VGPR v -> M = v + 8*(lane>=16); N = lane%16
  #pragma unroll
  for (int v = 0; v < 8; ++v) {
    int m = m0 + v + hi8;
    int x = ow0 + j0 + row;
    if (x < OW) {
      out[(((size_t)b * Cout + m) * OH + y) * OW + x] = acc[v] + bias[m];
    }
  }
}

extern "C" void kernel_launch(void* const* d_in, const int* in_sizes, int n_in,
                              void* d_out, int out_size, void* d_ws, size_t ws_size,
                              hipStream_t stream) {
  const float* input  = (const float*)d_in[0];
  const float* weight = (const float*)d_in[1];
  const float* bias   = (const float*)d_in[2];
  const int*   selh   = (const int*)d_in[3];
  const int*   selw   = (const int*)d_in[4];
  float* out = (float*)d_out;

  __bf16* wsW = (__bf16*)d_ws;    // 64*576*2 = 73,728 B of workspace

  // pre-pass: convert + permute weights (re-done every call; deterministic)
  hipLaunchKernelGGL(weight_permute_bf16, dim3((64 * 576 + 255) / 256), dim3(256),
                     0, stream, weight, wsW);

  dim3 grid(2, 63, 16);   // (ow tiles of 32, oh, batch)
  dim3 block(256);        // 8 waves: 4 M-tiles x 2 N-tiles of 16x16
  hipLaunchKernelGGL(sconv2d_wmma_kernel, grid, block, 0, stream,
                     input, wsW, bias, selh, selw, out);
}
